// Memory_41016937676880
// MI455X (gfx1250) — compile-verified, matrix operand order checked by reading, not run on hardware
//
#include <hip/hip_runtime.h>

#define B_ 256
#define N_ 65536
#define D_ 2048
#define K_ 16
#define INV_TEMP 20.0f
#define LAMB 0.05f
#define NT 4   // N-tiles (of 16) per workgroup

typedef __attribute__((ext_vector_type(16))) __bf16    v16bf;
typedef __attribute__((ext_vector_type(8)))  float     v8f;
typedef __attribute__((ext_vector_type(8)))  unsigned  v8u;

// round-to-nearest-even fp32 -> bf16, packed pair
__device__ __forceinline__ unsigned pack_bf16(float lo, float hi) {
  unsigned a = __builtin_bit_cast(unsigned, lo);
  unsigned b = __builtin_bit_cast(unsigned, hi);
  a = (a + 0x7FFFu + ((a >> 16) & 1u)) >> 16;
  b = (b + 0x7FFFu + ((b >> 16) & 1u)) >> 16;
  return a | (b << 16);
}

__device__ __forceinline__ float block_reduce(float v, float* red, int tid) {
  red[tid] = v;
  __syncthreads();
  for (int off = 128; off > 0; off >>= 1) {
    if (tid < off) red[tid] += red[tid + off];
    __syncthreads();
  }
  float r = red[0];
  __syncthreads();
  return r;
}

// Kernel 1: normalize x, emit packed-bf16 x, seed S[b] with pos + hard-neg terms.
__global__ __launch_bounds__(256) void prep_kernel(
    const float* __restrict__ inputs,   // [B,D]
    const int*   __restrict__ targets,  // [B]
    const float* __restrict__ pos_proto,// [N,D]
    const float* __restrict__ nh,       // [B,K,D]
    unsigned*    __restrict__ xpk,      // [B, D/2] packed bf16
    float*       __restrict__ S,        // [B]
    float*       __restrict__ LP) {     // [B]
  __shared__ float red[256];
  const int b = blockIdx.x;
  const int tid = threadIdx.x;
  const float* row = inputs + (size_t)b * D_;

  float ss = 0.f;
  for (int i = tid; i < D_; i += 256) { float v = row[i]; ss += v * v; }
  float inv = rsqrtf(block_reduce(ss, red, tid));

  // packed bf16 normalized x
  for (int p = tid; p < D_ / 2; p += 256)
    xpk[(size_t)b * (D_ / 2) + p] = pack_bf16(row[2 * p] * inv, row[2 * p + 1] * inv);

  // positive similarity
  const int t = targets[b];
  const float* pp = pos_proto + (size_t)t * D_;
  float lp = 0.f;
  for (int i = tid; i < D_; i += 256) lp += row[i] * inv * pp[i];
  lp = block_reduce(lp, red, tid);

  // hard negatives
  float hard = 0.f;
  for (int k = 0; k < K_; ++k) {
    const float* nr = nh + ((size_t)b * K_ + k) * D_;
    float s = 0.f;
    for (int i = tid; i < D_; i += 256) s += row[i] * inv * nr[i];
    s = block_reduce(s, red, tid);
    if (tid == 0) hard += expf(s * INV_TEMP);
  }
  if (tid == 0) {
    S[b]  = expf(lp * INV_TEMP) + LAMB * hard;  // init: pos term + hard-neg term
    LP[b] = lp * INV_TEMP;                      // log of l_pos
  }
}

// Kernel 2: bf16 WMMA GEMM x[256,2048] * neg[65536,2048]^T with fused
// exp(./TEMP) row-sum (target column excluded) accumulated into S via atomics.
__global__ __launch_bounds__(256) void gemm_kernel(
    const unsigned* __restrict__ xpk,   // [B, D/2] packed bf16
    const float*    __restrict__ negp,  // [N, D] fp32
    const int*      __restrict__ targets,
    float*          __restrict__ S) {
  __shared__ unsigned bstage[NT * 16 * 16];  // flat: (local_row*16 + kpair), bf16x2

  const int tid  = threadIdx.x;
  const int lane = tid & 31;
  const int wave = tid >> 5;           // 0..7, covers m-tiles {2w, 2w+1}
  const int half = lane >> 4;          // 0: K 0-7/16-23 ; 1: K 8-15/24-31
  const int l15  = lane & 15;
  const int nbase_wg = blockIdx.x * (NT * 16);

  v8f acc[2][NT];
  for (int mt = 0; mt < 2; ++mt)
    for (int nt = 0; nt < NT; ++nt)
      acc[mt][nt] = (v8f){0.f, 0.f, 0.f, 0.f, 0.f, 0.f, 0.f, 0.f};

  const int o0 = half ? 4 : 0;  // pair offset within each 16-K half

  // staging geometry (constant across K-steps): 64 rows x 8 quads of floats
  const int q0 = tid;            // quad ids 0..255
  const int q1 = tid + 256;      // quad ids 256..511
  const int r0 = q0 >> 3, c0 = (q0 & 7) * 4;
  const int r1 = q1 >> 3, c1 = (q1 & 7) * 4;
  const float* srow0 = negp + (size_t)(nbase_wg + r0) * D_ + c0;
  const float* srow1 = negp + (size_t)(nbase_wg + r1) * D_ + c1;

  for (int kb = 0; kb < D_; kb += 32) {
    // ---- stage neg 64x32 fp32 slice into LDS as packed bf16 (once per WG).
    // Two independent b128 loads in flight per thread, no exec-masked loop.
    {
      const float4 va = *(const float4*)(srow0 + kb);
      const float4 vb = *(const float4*)(srow1 + kb);
      if (kb + 32 < D_) {
        __builtin_prefetch(srow0 + kb + 32, 0, 3);  // global_prefetch_b8
        __builtin_prefetch(srow1 + kb + 32, 0, 3);
      }
      bstage[q0 * 2 + 0] = pack_bf16(va.x, va.y);
      bstage[q0 * 2 + 1] = pack_bf16(va.z, va.w);
      bstage[q1 * 2 + 0] = pack_bf16(vb.x, vb.y);
      bstage[q1 * 2 + 1] = pack_bf16(vb.z, vb.w);
    }
    __syncthreads();

    // ---- A fragments: row M = lane&15, K pairs {o0..o0+3} and {8+o0..} ----
    unsigned afr[2][8];
    for (int mt = 0; mt < 2; ++mt) {
      const int mrow = (wave * 2 + mt) * 16 + l15;
      const unsigned* xr = xpk + (size_t)mrow * (D_ / 2) + (kb >> 1);
      #pragma unroll
      for (int j = 0; j < 4; ++j) { afr[mt][j] = xr[o0 + j]; afr[mt][4 + j] = xr[8 + o0 + j]; }
    }
    // ---- B fragments from LDS: col N = lane&15, same K pattern ----
    unsigned bfr[NT][8];
    #pragma unroll
    for (int nt = 0; nt < NT; ++nt) {
      const unsigned* bp = &bstage[nt * 256 + l15 * 16];
      #pragma unroll
      for (int j = 0; j < 4; ++j) { bfr[nt][j] = bp[o0 + j]; bfr[nt][4 + j] = bp[8 + o0 + j]; }
    }

    #pragma unroll
    for (int mt = 0; mt < 2; ++mt) {
      v8u au;
      #pragma unroll
      for (int j = 0; j < 8; ++j) au[j] = afr[mt][j];
      const v16bf a = __builtin_bit_cast(v16bf, au);
      #pragma unroll
      for (int nt = 0; nt < NT; ++nt) {
        v8u bu;
        #pragma unroll
        for (int j = 0; j < 8; ++j) bu[j] = bfr[nt][j];
        const v16bf bb = __builtin_bit_cast(v16bf, bu);
        acc[mt][nt] = __builtin_amdgcn_wmma_f32_16x16x32_bf16(
            false, a, false, bb, (short)0, acc[mt][nt], false, false);
      }
    }
    __syncthreads();
  }

  // ---- epilogue: exp(sim/T), mask target column, per-row reduce, atomic add ----
  for (int mt = 0; mt < 2; ++mt) {
    const int mbase = (wave * 2 + mt) * 16;
    #pragma unroll
    for (int v = 0; v < 8; ++v) {
      const int row = mbase + half * 8 + v;       // C layout: M = v + 8*half
      const int tgt = targets[row];
      float s = 0.f;
      #pragma unroll
      for (int nt = 0; nt < NT; ++nt) {
        const int col = nbase_wg + nt * 16 + l15; // C layout: N = lane&15
        const float val = acc[mt][nt][v];
        s += (col == tgt) ? 0.f : expf(val * INV_TEMP);
      }
      // reduce across the 16 lanes holding this row (xor masks stay in-half)
      s += __shfl_xor(s, 1, 32);
      s += __shfl_xor(s, 2, 32);
      s += __shfl_xor(s, 4, 32);
      s += __shfl_xor(s, 8, 32);
      if (l15 == 0) atomicAdd(&S[row], s);        // global_atomic_add_f32
    }
  }
}

// Kernel 3: loss = -mean(LP[b] - log S[b])
__global__ __launch_bounds__(256) void loss_kernel(
    const float* __restrict__ S, const float* __restrict__ LP,
    float* __restrict__ out) {
  __shared__ float red[256];
  const int tid = threadIdx.x;
  float v = LP[tid] - logf(S[tid]);
  v = block_reduce(v, red, tid);
  if (tid == 0) out[0] = -v / (float)B_;
}

extern "C" void kernel_launch(void* const* d_in, const int* in_sizes, int n_in,
                              void* d_out, int out_size, void* d_ws, size_t ws_size,
                              hipStream_t stream) {
  const float* inputs  = (const float*)d_in[0];
  const int*   targets = (const int*)d_in[1];
  const float* posp    = (const float*)d_in[2];
  const float* negp    = (const float*)d_in[3];
  const float* nh      = (const float*)d_in[4];
  float* out = (float*)d_out;

  // workspace layout: packed-bf16 x (1 MB) | S (256 f32) | LP (256 f32)
  unsigned* xpk = (unsigned*)d_ws;
  float* S  = (float*)((char*)d_ws + (size_t)B_ * (D_ / 2) * sizeof(unsigned));
  float* LP = S + B_;

  prep_kernel<<<B_, 256, 0, stream>>>(inputs, targets, posp, nh, xpk, S, LP);
  gemm_kernel<<<N_ / (NT * 16), 256, 0, stream>>>(xpk, negp, targets, S);
  loss_kernel<<<1, 256, 0, stream>>>(S, LP, out);
}
